// DualEncoderBiLSTM_53077205844742
// MI455X (gfx1250) — matrix operand order repaired and checked by reading, not compiled, not run on hardware
//
#include <hip/hip_runtime.h>
#include <cstdint>
#include <cstddef>

typedef __attribute__((ext_vector_type(16))) __bf16 v16bf;
typedef __attribute__((ext_vector_type(8)))  float  v8f;

#define B_   256
#define T_   64
#define E_   300
#define H_   512
#define KX_  320          // x part of K, padded 300 -> 320 (multiple of 32)
#define KH_  512          // h part of K
#define KK_  832          // total padded K
#define NG_  2048         // 4*H gate columns
#define FORGET_BIAS_ 2.0f

struct CellPtrs {
  const float* bias[4];
  const float* wci[4];
  const float* wcf[4];
  const float* wco[4];
  const int*   len[4];
  const int*   ids[4];
};

__device__ __forceinline__ float sigf(float x) { return 1.0f / (1.0f + __expf(-x)); }

// gfx1250 async global->LDS copy: 16B per lane, INST_OFFSET applies to both
// the LDS destination and the global source address (ISA 08, §4.4).
__device__ __forceinline__ void async_b128(unsigned lds_off, const void* gaddr) {
  asm volatile("global_load_async_to_lds_b128 %0, %1, off"
               :: "v"(lds_off), "v"(gaddr) : "memory");
}
__device__ __forceinline__ void async_b128_o16(unsigned lds_off, const void* gaddr) {
  asm volatile("global_load_async_to_lds_b128 %0, %1, off offset:16"
               :: "v"(lds_off), "v"(gaddr) : "memory");
}
__device__ __forceinline__ void wait_async0() {
  asm volatile("s_wait_asynccnt 0x0" ::: "memory");
}

// ---------------------------------------------------------------------------
// Gather embeddings (with per-length sequence reversal for bw cells), cast to
// bf16, pad K 300->320. Layout: xs[cell][t][b][KX_]
// ---------------------------------------------------------------------------
__global__ __launch_bounds__(128) void prep_xs_kernel(const float* __restrict__ emb,
                                                      CellPtrs cp,
                                                      __bf16* __restrict__ xs) {
  int cell = blockIdx.y;
  int bt = blockIdx.x;
  int b = bt / T_, t = bt % T_;
  int len = cp.len[cell][b];
  int ts = t;
  if (cell & 1) ts = (t < len) ? (len - 1 - t) : t;   // _reverse_seq
  int id = cp.ids[cell][b * T_ + ts];
  const float* src = emb + (size_t)id * E_;
  __bf16* dst = xs + (((size_t)cell * T_ + t) * B_ + b) * KX_;
  for (int k = threadIdx.x; k < KX_; k += 128) {
    float v = (k < E_) ? src[k] : 0.0f;
    dst[k] = (__bf16)v;
  }
}

// ---------------------------------------------------------------------------
// Transpose + pad weights to bf16: wt[cell][n][k], k contiguous (N x K).
// ---------------------------------------------------------------------------
__global__ __launch_bounds__(128) void prep_w_kernel(const float* __restrict__ k0,
                                                     const float* __restrict__ k1,
                                                     const float* __restrict__ k2,
                                                     const float* __restrict__ k3,
                                                     __bf16* __restrict__ wt) {
  int cell = blockIdx.y;
  int n = blockIdx.x;
  const float* kern = cell == 0 ? k0 : cell == 1 ? k1 : cell == 2 ? k2 : k3;
  __bf16* dst = wt + ((size_t)cell * NG_ + n) * KK_;
  for (int k = threadIdx.x; k < KK_; k += 128) {
    float v = 0.0f;
    if (k < E_)        v = kern[(size_t)k * NG_ + n];
    else if (k >= KX_) v = kern[(size_t)(k - (KX_ - E_)) * NG_ + n];
    dst[k] = (__bf16)v;
  }
}

__global__ void init_state_kernel(float* hs, float* cs, __bf16* hbf) {
  size_t i = (size_t)blockIdx.x * blockDim.x + threadIdx.x;
  size_t nf = (size_t)4 * B_ * H_;
  if (i < nf) { hs[i] = 0.0f; cs[i] = 0.0f; }
  if (i < 2 * nf) hbf[i] = (__bf16)0.0f;
}

// ---------------------------------------------------------------------------
// One LSTM time step for all 4 cells, fully fused:
//   gates = [x_t | h_{t-1}] @ W  (bf16 WMMA, f32 acc) + cell-update epilogue.
// Double-buffered LDS tiles filled by async global->LDS copies (ASYNCcnt),
// one barrier per K-iteration; K-loop unrolled x2 so accumulators stay pinned.
// ---------------------------------------------------------------------------
__global__ __launch_bounds__(128) void lstm_step_kernel(
    const __bf16* __restrict__ xs, const __bf16* __restrict__ wt,
    const __bf16* __restrict__ hbf_in, __bf16* __restrict__ hbf_out,
    float* __restrict__ hstate, float* __restrict__ cstate,
    CellPtrs cp, int t) {
  // per buffer: A 64x32 bf16 (4KB) then B 4x16x32 bf16 (4KB)
  __shared__ int4 smem[2][512];

  int cell = blockIdx.z;
  int nb   = blockIdx.x;   // hidden tile 0..31
  int mblk = blockIdx.y;   // 0..3 (64 batch rows each)
  int tid  = threadIdx.x;
  int wave = tid >> 5, lane = tid & 31;
  int row  = lane & 15, sel = lane >> 4;
  int nbase = nb * 16;

  const __bf16* xbase = xs + (((size_t)cell * T_ + t) * B_ + mblk * 64) * KX_;
  const __bf16* hbase = hbf_in + ((size_t)cell * B_ + (size_t)mblk * 64) * KH_;
  const __bf16* wbase = wt + (size_t)cell * NG_ * KK_;

  int r = tid >> 1, half = tid & 1;          // A staging role: row r, 16-elem half
  int g = tid >> 5, nl = (tid >> 1) & 15;    // B staging role: gate g, local col nl

  // wave-relative LDS byte offsets (flat shared addr keeps offset in [31:0])
  unsigned lds0 = (unsigned)(uintptr_t)&smem[0][0];
  unsigned dstA = lds0 + (unsigned)((r * 32 + half * 16) * 2);
  unsigned dstB = lds0 + 4096u + (unsigned)(((g * 16 + nl) * 32 + half * 16) * 2);
  const __bf16* srcB0 = wbase + ((size_t)(g * H_ + nbase + nl)) * KK_ + half * 16;

  auto stage = [&](int kb, unsigned bufoff) {
    int kg = kb * 32 + half * 16;
    const __bf16* sA = (kb < 10) ? (xbase + (size_t)r * KX_ + kg)
                                 : (hbase + (size_t)r * KH_ + (kg - KX_));
    const __bf16* sB = srcB0 + kb * 32;
    async_b128(dstA + bufoff, sA);
    async_b128_o16(dstA + bufoff, sA);
    async_b128(dstB + bufoff, sB);
    async_b128_o16(dstB + bufoff, sB);
  };

  v8f acc[4] = {};
  union V16 { int4 q[2]; v16bf v; };

  auto compute = [&](int buf) {
    // A: lane = row M, contiguous 16 K per lane half
    V16 a;
    int ai = (wave * 16 + row) * 4 + sel * 2;
    a.q[0] = smem[buf][ai];
    a.q[1] = smem[buf][ai + 1];
    #pragma unroll
    for (int gg = 0; gg < 4; ++gg) {
      V16 bb;  // B: lane = col N, contiguous 16 K per lane half
      int bi = 256 + (gg * 16 + row) * 4 + sel * 2;
      bb.q[0] = smem[buf][bi];
      bb.q[1] = smem[buf][bi + 1];
      acc[gg] = __builtin_amdgcn_wmma_f32_16x16x32_bf16(
          false, a.v, false, bb.v, (short)0, acc[gg], false, false);
    }
  };

  stage(0, 0);
  wait_async0();
  __syncthreads();

  for (int kb = 0; kb < 26; kb += 2) {       // K = 832 = 26 * 32, 26 even
    stage(kb + 1, 8192u);                    // prefetch odd tile into buf1
    compute(0);
    wait_async0();
    __syncthreads();

    if (kb + 2 < 26) stage(kb + 2, 0u);      // prefetch next even tile into buf0
    compute(1);
    wait_async0();
    __syncthreads();
  }

  // ---- fused cell update epilogue. C layout: VGPR v -> M = v + 8*(lane>>4),
  //      N = lane&15.
  int k = nbase + row;
  float b_i = cp.bias[cell][0 * H_ + k];
  float b_j = cp.bias[cell][1 * H_ + k];
  float b_f = cp.bias[cell][2 * H_ + k];
  float b_o = cp.bias[cell][3 * H_ + k];
  float wci = cp.wci[cell][k], wcf = cp.wcf[cell][k], wco = cp.wco[cell][k];
  int mbase = mblk * 64 + wave * 16;
  #pragma unroll
  for (int vi = 0; vi < 8; ++vi) {
    int m = mbase + vi + sel * 8;
    size_t idx = ((size_t)cell * B_ + m) * H_ + k;
    float c  = cstate[idx];
    float hp = hstate[idx];
    float iv = acc[0][vi] + b_i;
    float jv = acc[1][vi] + b_j;
    float fv = acc[2][vi] + b_f;
    float ov = acc[3][vi] + b_o;
    float ig = sigf(iv + wci * c);
    float fg = sigf(fv + wcf * c + FORGET_BIAS_);
    float cn = fg * c + ig * tanhf(jv);
    float og = sigf(ov + wco * cn);
    float hn = og * tanhf(cn);
    bool msk = t < cp.len[cell][m];
    float co = msk ? cn : c;
    float ho = msk ? hn : hp;
    cstate[idx] = co;
    hstate[idx] = ho;
    hbf_out[idx] = (__bf16)ho;   // feeds next step's GEMM
  }
}

// ---------------------------------------------------------------------------
// enc = concat(h_fw, h_bw) cast to bf16 (row-major B x 1024)
// ---------------------------------------------------------------------------
__global__ __launch_bounds__(256) void cast_enc_kernel(const float* __restrict__ hstate,
                                                       __bf16* __restrict__ encq,
                                                       __bf16* __restrict__ encr) {
  int b = blockIdx.x;
  int side = blockIdx.y;  // 0 = query (cells 0,1), 1 = reply (cells 2,3)
  __bf16* dst = (side ? encr : encq) + (size_t)b * 1024;
  int cbase = side * 2;
  for (int k = threadIdx.x; k < 1024; k += 256) {
    int cell = cbase + (k >> 9);
    int kk = k & 511;
    dst[k] = (__bf16)hstate[((size_t)cell * B_ + b) * H_ + kk];
  }
}

// ---------------------------------------------------------------------------
// distances = encq @ encr^T  (256 x 256, K=1024) via bf16 WMMA
// ---------------------------------------------------------------------------
__global__ __launch_bounds__(32) void distances_kernel(const __bf16* __restrict__ encq,
                                                       const __bf16* __restrict__ encr,
                                                       float* __restrict__ out) {
  int lane = threadIdx.x;
  int row = lane & 15, sel = lane >> 4;
  int mbase = blockIdx.y * 16, nbase = blockIdx.x * 16;
  const __bf16* arow = encq + (size_t)(mbase + row) * 1024;
  const __bf16* brow = encr + (size_t)(nbase + row) * 1024;
  v8f acc = {};
  union V16 { int4 q[2]; v16bf v; };
  for (int kb = 0; kb < 32; ++kb) {
    V16 a, b;
    a.q[0] = *(const int4*)(arow + kb * 32 + sel * 16);
    a.q[1] = *(const int4*)(arow + kb * 32 + sel * 16 + 8);
    b.q[0] = *(const int4*)(brow + kb * 32 + sel * 16);
    b.q[1] = *(const int4*)(brow + kb * 32 + sel * 16 + 8);
    acc = __builtin_amdgcn_wmma_f32_16x16x32_bf16(
        false, a.v, false, b.v, (short)0, acc, false, false);
  }
  #pragma unroll
  for (int vi = 0; vi < 8; ++vi)
    out[(size_t)(mbase + vi + sel * 8) * 256 + nbase + row] = acc[vi];
}

// ---------------------------------------------------------------------------
extern "C" void kernel_launch(void* const* d_in, const int* in_sizes, int n_in,
                              void* d_out, int out_size, void* d_ws, size_t ws_size,
                              hipStream_t stream) {
  (void)in_sizes; (void)n_in; (void)out_size; (void)ws_size;

  const int*   q_ids = (const int*)d_in[0];
  const int*   r_ids = (const int*)d_in[1];
  const int*   q_len = (const int*)d_in[2];
  const int*   r_len = (const int*)d_in[3];
  const float* emb   = (const float*)d_in[4];

  const float* kern[4];
  CellPtrs cp;
  for (int c = 0; c < 4; ++c) {           // order: q_fw, q_bw, r_fw, r_bw
    int base = 5 + c * 5;                 // kernel, bias, wci, wcf, wco
    kern[c]    = (const float*)d_in[base + 0];
    cp.bias[c] = (const float*)d_in[base + 1];
    cp.wci[c]  = (const float*)d_in[base + 2];
    cp.wcf[c]  = (const float*)d_in[base + 3];
    cp.wco[c]  = (const float*)d_in[base + 4];
  }
  cp.len[0] = q_len; cp.len[1] = q_len; cp.len[2] = r_len; cp.len[3] = r_len;
  cp.ids[0] = q_ids; cp.ids[1] = q_ids; cp.ids[2] = r_ids; cp.ids[3] = r_ids;

  // workspace carve (~63 MB)
  char* ws = (char*)d_ws;
  size_t off = 0;
  __bf16* xs   = (__bf16*)(ws + off); off += (size_t)4 * T_ * B_ * KX_ * 2;
  __bf16* wt   = (__bf16*)(ws + off); off += (size_t)4 * NG_ * KK_ * 2;
  float*  hs   = (float*)(ws + off);  off += (size_t)4 * B_ * H_ * 4;
  float*  cs   = (float*)(ws + off);  off += (size_t)4 * B_ * H_ * 4;
  __bf16* hbf  = (__bf16*)(ws + off); off += (size_t)2 * 4 * B_ * H_ * 2;
  __bf16* encq = (__bf16*)(ws + off); off += (size_t)B_ * 1024 * 2;
  __bf16* encr = (__bf16*)(ws + off);

  prep_xs_kernel<<<dim3(B_ * T_, 4), 128, 0, stream>>>(emb, cp, xs);
  prep_w_kernel<<<dim3(NG_, 4), 128, 0, stream>>>(kern[0], kern[1], kern[2], kern[3], wt);
  init_state_kernel<<<(2 * 4 * B_ * H_ + 255) / 256, 256, 0, stream>>>(hs, cs, hbf);

  size_t hbuf = (size_t)4 * B_ * H_;
  for (int t = 0; t < T_; ++t) {
    const __bf16* hin = hbf + (size_t)(t & 1) * hbuf;
    __bf16* hout      = hbf + (size_t)((t + 1) & 1) * hbuf;
    lstm_step_kernel<<<dim3(32, 4, 4), 128, 0, stream>>>(xs, wt, hin, hout, hs, cs, cp, t);
  }

  cast_enc_kernel<<<dim3(B_, 2), 256, 0, stream>>>(hs, encq, encr);
  distances_kernel<<<dim3(16, 16), 32, 0, stream>>>(encq, encr, (float*)d_out);
}
